// EnhancedHeterogeneousGNN_2989297238398
// MI455X (gfx1250) — matrix operand order repaired and checked by reading, not compiled, not run on hardware
//
#include <hip/hip_runtime.h>
#include <math.h>

typedef __attribute__((ext_vector_type(16))) _Float16 v16h;
typedef __attribute__((ext_vector_type(8)))  float    v8f;

constexpr int cB = 4, cU = 16, cT = 64, cE = 128, cH = 4, cHD = 32;

// ---------------------------------------------------------------------------
// WMMA fragment loaders (fp32 memory -> f16 fragments).
// Layouts per CDNA5 ISA 7.12.2 (wave32):
//  A 16x32 f16: lane m=lane&15, kh=lane>>4; VGPR v holds K = 16*(v>=4)+8*kh+2*(v&3)+{0,1}
//  B 32x16 f16: lane n=lane&15, kh=lane>>4; VGPR v holds K = 16*(v>=4)+8*kh+2*(v&3)+{0,1}
//  C/D 16x16 f32: lane n=lane&15; VGPR r holds M = r + 8*(lane>>4)
// ---------------------------------------------------------------------------
__device__ __forceinline__ v16h cvt16(float4 q0, float4 q1, float4 q2, float4 q3) {
  v16h f;
  f[0]  = (_Float16)q0.x; f[1]  = (_Float16)q0.y;
  f[2]  = (_Float16)q0.z; f[3]  = (_Float16)q0.w;
  f[4]  = (_Float16)q1.x; f[5]  = (_Float16)q1.y;
  f[6]  = (_Float16)q1.z; f[7]  = (_Float16)q1.w;
  f[8]  = (_Float16)q2.x; f[9]  = (_Float16)q2.y;
  f[10] = (_Float16)q2.z; f[11] = (_Float16)q2.w;
  f[12] = (_Float16)q3.x; f[13] = (_Float16)q3.y;
  f[14] = (_Float16)q3.z; f[15] = (_Float16)q3.w;
  return f;
}

// Fully in-bounds, ld % 4 == 0: 4x global_load_b128, no predication.
__device__ __forceinline__ v16h frag_a_fast(const float* __restrict__ A, int row0,
                                            int ld, int k0) {
  const int lane = threadIdx.x & 31;
  const float* p = A + (size_t)(row0 + (lane & 15)) * ld + k0 + (lane >> 4) * 8;
  return cvt16(*(const float4*)(p),      *(const float4*)(p + 4),
               *(const float4*)(p + 16), *(const float4*)(p + 20));
}

__device__ __forceinline__ v16h frag_b_fast(const float* __restrict__ W, int col0,
                                            int ld, int k0) {
  const int lane = threadIdx.x & 31;
  const float* p = W + (size_t)(col0 + (lane & 15)) * ld + k0 + (lane >> 4) * 8;
  return cvt16(*(const float4*)(p),      *(const float4*)(p + 4),
               *(const float4*)(p + 16), *(const float4*)(p + 20));
}

// Safe path: clamped (always-valid) addresses, unconditional loads,
// multiply-by-mask instead of selecting loaded values (no EXEC regions).
__device__ __forceinline__ v16h frag_a_safe(const float* __restrict__ A, int row0,
                                            int nrows, int kdim, int ld, int k0) {
  const int lane = threadIdx.x & 31;
  const int r = row0 + (lane & 15), kh = lane >> 4;
  const int rc = (r < nrows) ? r : (nrows - 1);
  const float rmask = (r < nrows) ? 1.f : 0.f;
  const float* p = A + (size_t)rc * ld;
  v16h f;
#pragma unroll
  for (int v = 0; v < 8; ++v) {
    const int k = k0 + ((v & 4) << 2) + kh * 8 + (v & 3) * 2;
    const int kc0 = (k < kdim) ? k : 0;
    const int kc1 = (k + 1 < kdim) ? (k + 1) : 0;
    const float m0 = (k < kdim) ? rmask : 0.f;
    const float m1 = (k + 1 < kdim) ? rmask : 0.f;
    f[2 * v]     = (_Float16)(p[kc0] * m0);
    f[2 * v + 1] = (_Float16)(p[kc1] * m1);
  }
  return f;
}

__device__ __forceinline__ v16h frag_b_safe(const float* __restrict__ W, int col0,
                                            int ncols, int kdim, int ld, int k0) {
  const int lane = threadIdx.x & 31;
  const int r = col0 + (lane & 15), kh = lane >> 4;
  const int rc = (r < ncols) ? r : (ncols - 1);
  const float rmask = (r < ncols) ? 1.f : 0.f;
  const float* p = W + (size_t)rc * ld;
  v16h f;
#pragma unroll
  for (int v = 0; v < 8; ++v) {
    const int k = k0 + ((v & 4) << 2) + kh * 8 + (v & 3) * 2;
    const int kc0 = (k < kdim) ? k : 0;
    const int kc1 = (k + 1 < kdim) ? (k + 1) : 0;
    const float m0 = (k < kdim) ? rmask : 0.f;
    const float m1 = (k + 1 < kdim) ? rmask : 0.f;
    f[2 * v]     = (_Float16)(p[kc0] * m0);
    f[2 * v + 1] = (_Float16)(p[kc1] * m1);
  }
  return f;
}

__device__ __forceinline__ float gelu_f(float x) {
  return 0.5f * x * (1.0f + erff(x * 0.7071067811865476f));
}

// ---------------------------------------------------------------------------
// Generic GEMM: C(N,Dout) = act(A(N,K) @ W(Dout,K)^T + bias) [+ resid]
// tstore!=0 => C[col*ldt + row] (transposed store, used to build V^T)
// One wave = one 16x16 tile.  Uniform fast/slow split hoisted out of K-loop.
// ---------------------------------------------------------------------------
__global__ void gemm_k(const float* __restrict__ A, const float* __restrict__ W,
                       const float* __restrict__ bias, const float* __restrict__ resid,
                       float* __restrict__ C, int N, int K, int Dout,
                       int act, int tstore, int ldt)
{
  const int row0 = blockIdx.x * 16;
  const int col0 = blockIdx.y * 16;
  v8f acc = {};
  if (((K & 31) == 0) & (row0 + 16 <= N) & (col0 + 16 <= Dout)) {
    for (int k0 = 0; k0 < K; k0 += 32) {
      v16h a = frag_a_fast(A, row0, K, k0);
      v16h b = frag_b_fast(W, col0, K, k0);
      acc = __builtin_amdgcn_wmma_f32_16x16x32_f16(false, a, false, b, (short)0,
                                                   acc, false, false);
    }
  } else {
    for (int k0 = 0; k0 < K; k0 += 32) {
      v16h a = frag_a_safe(A, row0, N, K, K, k0);
      v16h b = frag_b_safe(W, col0, Dout, K, K, k0);
      acc = __builtin_amdgcn_wmma_f32_16x16x32_f16(false, a, false, b, (short)0,
                                                   acc, false, false);
    }
  }
  const int lane = threadIdx.x & 31;
  const int n = lane & 15, kh = lane >> 4;
  const int col = col0 + n;
  if (col >= Dout) return;
  const float bv = bias ? bias[col] : 0.f;
#pragma unroll
  for (int r = 0; r < 8; ++r) {
    int rowm = row0 + r + kh * 8;
    if (rowm < N) {
      float v = acc[r] + bv;
      if (act == 1) v = gelu_f(v);
      else if (act == 2) v = tanhf(v);
      if (resid) v += resid[(size_t)rowm * Dout + col];
      if (tstore) C[(size_t)col * ldt + rowm] = v;
      else        C[(size_t)rowm * Dout + col] = v;
    }
  }
}

// ---------------------------------------------------------------------------
// Attention scores (one batch): S[h,n,m] = (Q[n]·K[m]) * scale into padded
// (N, Mp) rows; pad columns and masked entries get -inf.  N,Mp tile-exact.
// ---------------------------------------------------------------------------
__global__ void scores_k(const float* __restrict__ Q, const float* __restrict__ Km,
                         const float* __restrict__ mask, float* __restrict__ S,
                         int N, int M, int Mp, float scale)
{
  const int col0 = blockIdx.x * 16;
  const int row0 = blockIdx.y * 16;
  const int h = blockIdx.z;
  v16h a = frag_a_fast(Q + h * cHD, row0, cE, 0);
  v16h b = frag_b_fast(Km + h * cHD, col0, cE, 0);
  v8f acc = {};
  acc = __builtin_amdgcn_wmma_f32_16x16x32_f16(false, a, false, b, (short)0,
                                               acc, false, false);
  const int lane = threadIdx.x & 31;
  const int n = lane & 15, kh = lane >> 4;
  const int col = col0 + n;
#pragma unroll
  for (int r = 0; r < 8; ++r) {
    const int rowm = row0 + r + kh * 8;
    float v = -INFINITY;
    if (col < M) {
      v = acc[r] * scale;
      if (mask && mask[(size_t)rowm * M + col] == 0.f) v = -INFINITY;
    }
    S[((size_t)h * N + rowm) * Mp + col] = v;
  }
}

// ---------------------------------------------------------------------------
// Attention output (one batch): O[n, h*HD+d] = sum_m P[h,n,m] * V^T[h*HD+d, m]
// P is (N, Mp) padded; pad probabilities are exactly 0.  Fully fast path.
// ---------------------------------------------------------------------------
__global__ void attnout_k(const float* __restrict__ S, const float* __restrict__ VT,
                          float* __restrict__ O, int N, int Mp, int ldv)
{
  const int col0 = blockIdx.x * 16;      // d within head (grid.x = 2)
  const int row0 = blockIdx.y * 16;
  const int h = blockIdx.z;
  const float* P  = S  + (size_t)h * N * Mp;
  const float* Vh = VT + (size_t)h * cHD * ldv;
  v8f acc = {};
  for (int k0 = 0; k0 < Mp; k0 += 32) {
    v16h a = frag_a_fast(P, row0, Mp, k0);
    v16h b = frag_b_fast(Vh, col0, ldv, k0);
    acc = __builtin_amdgcn_wmma_f32_16x16x32_f16(false, a, false, b, (short)0,
                                                 acc, false, false);
  }
  const int lane = threadIdx.x & 31;
  const int n = lane & 15, kh = lane >> 4;
  const int col = col0 + n;
#pragma unroll
  for (int r = 0; r < 8; ++r) {
    const int rowm = row0 + r + kh * 8;
    O[(size_t)rowm * cE + h * cHD + col] = acc[r];
  }
}

// ---------------------------------------------------------------------------
// Row softmax, in place.  One block (256 thr) per row of length M (padded).
// ---------------------------------------------------------------------------
__global__ void softmax_k(float* __restrict__ S, int M)
{
  float* row = S + (size_t)blockIdx.x * M;
  __shared__ float red[256];
  const int tid = threadIdx.x;
  float mx = -INFINITY;
  for (int i = tid; i < M; i += 256) mx = fmaxf(mx, row[i]);
  red[tid] = mx; __syncthreads();
  for (int s = 128; s > 0; s >>= 1) { if (tid < s) red[tid] = fmaxf(red[tid], red[tid + s]); __syncthreads(); }
  mx = red[0]; __syncthreads();
  float sm = 0.f;
  for (int i = tid; i < M; i += 256) { float e = expf(row[i] - mx); row[i] = e; sm += e; }
  red[tid] = sm; __syncthreads();
  for (int s = 128; s > 0; s >>= 1) { if (tid < s) red[tid] += red[tid + s]; __syncthreads(); }
  const float inv = 1.f / red[0];
  for (int i = tid; i < M; i += 256) row[i] *= inv;
}

// ---------------------------------------------------------------------------
// LayerNorm: one block of F threads per row (F = 128 or 256, power of 2)
// ---------------------------------------------------------------------------
__global__ void ln_k(const float* __restrict__ X, const float* __restrict__ g,
                     const float* __restrict__ bta, float* __restrict__ Y, int F)
{
  __shared__ float red[256];
  const int tid = threadIdx.x;
  const size_t base = (size_t)blockIdx.x * F;
  const float x = X[base + tid];
  red[tid] = x; __syncthreads();
  for (int s = blockDim.x / 2; s > 0; s >>= 1) { if (tid < s) red[tid] += red[tid + s]; __syncthreads(); }
  const float m = red[0] / (float)F; __syncthreads();
  const float d = x - m;
  red[tid] = d * d; __syncthreads();
  for (int s = blockDim.x / 2; s > 0; s >>= 1) { if (tid < s) red[tid] += red[tid + s]; __syncthreads(); }
  const float var = red[0] / (float)F;
  Y[base + tid] = d * rsqrtf(var + 1e-5f) * g[tid] + bta[tid];
}

// ---------------------------------------------------------------------------
// Sinusoidal positional encoding (PE table computed analytically).
// ---------------------------------------------------------------------------
__global__ void posenc_k(float* __restrict__ X, const float* __restrict__ feat, int fin)
{
  const int row = blockIdx.x, e = threadIdx.x;
  const float p0 = feat[(size_t)row * fin + 0];
  const float p1 = feat[(size_t)row * fin + 1];
  int i0 = (int)(p0 * 100.f); i0 = i0 < 0 ? 0 : (i0 > 199 ? 199 : i0);
  int i1 = (int)(p1 * 100.f); i1 = i1 < 0 ? 0 : (i1 > 199 ? 199 : i1);
  const float divv = expf(-(float)(e & ~1) * (9.210340371976184f / 128.f)); // ln(1e4)/E
  const float a0 = (float)i0 * divv, a1 = (float)i1 * divv;
  const float pe = (e & 1) ? (cosf(a0) + cosf(a1)) : (sinf(a0) + sinf(a1));
  X[(size_t)row * cE + e] += pe;
}

// ---------------------------------------------------------------------------
// kNN(5) adjacency mask from task positions (mask must be pre-zeroed)
// ---------------------------------------------------------------------------
__global__ void knn_mask_k(const float* __restrict__ tf, float* __restrict__ mask)
{
  const int idx = blockIdx.x * blockDim.x + threadIdx.x;
  if (idx >= cB * cT) return;
  const int b = idx / cT, n = idx % cT;
  const float px = tf[((size_t)b * cT + n) * 4 + 0];
  const float py = tf[((size_t)b * cT + n) * 4 + 1];
  unsigned long long chosen = 0ull;
  for (int j = 0; j < 5; ++j) {
    float best = INFINITY; int bi = 0;
    for (int m = 0; m < cT; ++m) {
      if ((chosen >> m) & 1ull) continue;
      const float dx = px - tf[((size_t)b * cT + m) * 4 + 0];
      const float dy = py - tf[((size_t)b * cT + m) * 4 + 1];
      const float d = sqrtf(dx * dx + dy * dy);
      if (d < best) { best = d; bi = m; }
    }
    chosen |= (1ull << bi);
    mask[((size_t)b * cT + n) * cT + bi] = 1.f;
  }
}

// Zero-padded copy: dst(rows, dcols) <- src(rows, scols), pad cols with 0
__global__ void padcopy_k(const float* __restrict__ src, float* __restrict__ dst,
                          int rows, int scols, int dcols)
{
  const int i = blockIdx.x * blockDim.x + threadIdx.x;
  if (i >= rows * dcols) return;
  const int c = i % dcols, r = i / dcols;
  dst[i] = (c < scols) ? src[(size_t)r * scols + c] : 0.f;
}

// tf = concat(broadcast(task_feat, U), edges) -> (B*U*T, Fp), zero padded
__global__ void tf_concat_k(const float* __restrict__ tfeat, const float* __restrict__ edges,
                            float* __restrict__ out, int fin, int Fp)
{
  const int tot = cB * cU * cT * Fp;
  const int i = blockIdx.x * blockDim.x + threadIdx.x;
  if (i >= tot) return;
  const int j = i % Fp; const int row = i / Fp;
  const int t = row % cT; const int bu = row / cT; const int b = bu / cU;
  float v = 0.f;
  if (j < fin)           v = tfeat[((size_t)b * cT + t) * fin + j];
  else if (j < fin + 3)  v = edges[(size_t)row * 3 + (j - fin)];
  out[i] = v;
}

// q_big[(bu*T + t), e] = u2[bu, e]
__global__ void bcast_q_k(const float* __restrict__ u2, float* __restrict__ qb)
{
  const int i = blockIdx.x * blockDim.x + threadIdx.x;
  if (i >= cB * cU * cT * cE) return;
  const int e = i & (cE - 1); const int row = i >> 7; const int bu = row / cT;
  qb[i] = u2[(size_t)bu * cE + e];
}

// mean over T: cm[bu, e] = mean_t cross[(bu*T+t), e]
__global__ void mean_t_k(const float* __restrict__ cr, float* __restrict__ cm)
{
  const int i = blockIdx.x * blockDim.x + threadIdx.x;
  if (i >= cB * cU * cE) return;
  const int e = i & (cE - 1); const int bu = i >> 7;
  float s = 0.f;
  for (int t = 0; t < cT; ++t) s += cr[((size_t)bu * cT + t) * cE + e];
  cm[i] = s * (1.f / (float)cT);
}

// attentive pooling: g[b, base+e] = sum_n softmax_n(sc)[n] * x[b*N+n, e]
__global__ void pool_k(const float* __restrict__ sc, const float* __restrict__ x,
                       float* __restrict__ g, int N, int base)
{
  const int b = blockIdx.x, e = threadIdx.x;
  const float* s = sc + (size_t)b * N;
  float mx = -INFINITY;
  for (int n = 0; n < N; ++n) mx = fmaxf(mx, s[n]);
  float Z = 0.f, acc = 0.f;
  for (int n = 0; n < N; ++n) {
    const float w = expf(s[n] - mx);
    Z += w;
    acc += w * x[((size_t)b * N + n) * cE + e];
  }
  g[(size_t)b * (2 * cE) + base + e] = acc / Z;
}

// ===========================================================================
// Host side
// ===========================================================================
struct LinW  { const float *b, *w; };
struct MhaW  { const float *ln_b, *ln_g, *wk, *wo_b, *wo_w, *wq, *wv; };
struct UsvW  { LinW ffn1, ffn2; const float *ln_b, *ln_g; MhaW self_attn, task_attn; LinW task_proj, usv_proj; };
struct TaskW { LinW ffn1, ffn2, ffn3; const float *ln_b, *ln_g; MhaW self_attn; LinW task_proj; MhaW usv_attn; LinW usv_proj; };

// Workspace layout (in floats)
constexpr size_t F_U    = 0;
constexpr size_t F_T    = F_U    + 8192;
constexpr size_t F_MASK = F_T    + 32768;
constexpr size_t F_UP   = F_MASK + 16384;
constexpr size_t F_U2   = F_UP   + 8192;
constexpr size_t F_TW   = F_U2   + 8192;
constexpr size_t F_T2   = F_TW   + 32768;
constexpr size_t F_Q    = F_T2   + 32768;
constexpr size_t F_K    = F_Q    + 524288;
constexpr size_t F_VT   = F_K    + 524288;
constexpr size_t F_O    = F_VT   + 524288;
constexpr size_t F_PRE  = F_O    + 524288;
constexpr size_t F_S    = F_PRE  + 524288;   // per-batch scores: H*1024*1024
constexpr size_t F_TF   = F_S    + 4194304;  // 4096 x 160 padded
constexpr size_t F_TP   = F_TF   + 655360;
constexpr size_t F_QBIG = F_TP   + 524288;
constexpr size_t F_CROSS= F_QBIG + 524288;
constexpr size_t F_CM   = F_CROSS+ 524288;
constexpr size_t F_F1   = F_CM   + 8192;
constexpr size_t F_F2   = F_F1   + 131072;
constexpr size_t F_PA   = F_F2   + 65536;
constexpr size_t F_PS   = F_PA   + 16384;
constexpr size_t F_GC   = F_PS   + 256;
constexpr size_t F_G1   = F_GC   + 1024;
constexpr size_t F_E1   = F_G1   + 1024;
constexpr size_t F_PRE2 = F_E1   + 1024;
constexpr size_t F_G    = F_PRE2 + 1024;
constexpr size_t F_PU0  = F_G    + 1024;     // usv_features padded 64x32
constexpr size_t F_PT0  = F_PU0  + 2048;     // task_features padded 256x32
constexpr size_t F_PWU0 = F_PT0  + 8192;     // usv[0].usv_proj.w padded 128x32
constexpr size_t F_PWT0 = F_PWU0 + 4096;     // task[0].task_proj.w padded 128x32
constexpr size_t F_PWTU = F_PWT0 + 4096;     // usv[i].task_proj.w padded 128x160 x6
constexpr size_t PWTU_STRIDE = 128 * 160;

static void launch_gemm(hipStream_t st, const float* A, const float* W, const float* bias,
                        const float* resid, float* C, int N, int K, int Dout,
                        int act, int tstore, int ldt)
{
  dim3 g((unsigned)((N + 15) / 16), (unsigned)((Dout + 15) / 16));
  gemm_k<<<g, 32, 0, st>>>(A, W, bias, resid, C, N, K, Dout, act, tstore, ldt);
}

// Full MHA + residual + layernorm.  q_in: (B*N, E), kv_in: (B*M, E).
static void run_mha(hipStream_t st, const MhaW& w, const float* q_in, int N,
                    const float* kv_in, int M, const float* mask, float* out, float* ws)
{
  float* Q  = ws + F_Q;  float* Kb = ws + F_K;  float* VT = ws + F_VT;
  float* O  = ws + F_O;  float* PRE = ws + F_PRE; float* S = ws + F_S;
  const int NT = cB * N, MT = cB * M;
  const int Mp = (M + 31) & ~31;                 // padded key dim
  launch_gemm(st, q_in,  w.wq, nullptr, nullptr, Q,  NT, cE, cE, 0, 0, 0);
  launch_gemm(st, kv_in, w.wk, nullptr, nullptr, Kb, MT, cE, cE, 0, 0, 0);
  launch_gemm(st, kv_in, w.wv, nullptr, nullptr, VT, MT, cE, cE, 0, 1, MT); // V^T (E, MT)
  const float scale = 0.17677669529663687f; // 1/sqrt(HD)
  for (int b = 0; b < cB; ++b) {
    dim3 gs((unsigned)(Mp / 16), (unsigned)(N / 16), cH);
    scores_k<<<gs, 32, 0, st>>>(Q + (size_t)b * N * cE, Kb + (size_t)b * M * cE,
                                mask ? mask + (size_t)b * N * M : nullptr, S, N, M, Mp, scale);
    softmax_k<<<cH * N, 256, 0, st>>>(S, Mp);
    dim3 go(2, (unsigned)(N / 16), cH);
    attnout_k<<<go, 32, 0, st>>>(S, VT + (size_t)b * M, O + (size_t)b * N * cE, N, Mp, MT);
  }
  launch_gemm(st, O, w.wo_w, w.wo_b, q_in, PRE, NT, cE, cE, 0, 0, 0);
  ln_k<<<NT, cE, 0, st>>>(PRE, w.ln_g, w.ln_b, out, cE);
}

// uA(Ku)/wu: padded usv_proj operands; posfeat(fpos): raw features for PE.
// tin(fin_t): raw t input for concat; wtp/Ftfp: padded task_proj weight/width.
static void usv_layer(hipStream_t st, const UsvW& w,
                      const float* uA, int Ku, const float* wu,
                      const float* posfeat, int fpos,
                      const float* tin, int fin_t, const float* wtp, int Ftfp,
                      const float* edges, float* ws, float* uout)
{
  float* UP = ws + F_UP;  float* U2 = ws + F_U2;  float* TF = ws + F_TF;
  float* TP = ws + F_TP;  float* QB = ws + F_QBIG; float* CR = ws + F_CROSS;
  float* CM = ws + F_CM;  float* FF1 = ws + F_F1;  float* PRE = ws + F_PRE;

  launch_gemm(st, uA, wu, w.usv_proj.b, nullptr, UP, cB * cU, Ku, cE, 0, 0, 0);
  posenc_k<<<cB * cU, cE, 0, st>>>(UP, posfeat, fpos);
  run_mha(st, w.self_attn, UP, cU, UP, cU, nullptr, U2, ws);

  const int tot = cB * cU * cT * Ftfp;
  tf_concat_k<<<(tot + 255) / 256, 256, 0, st>>>(tin, edges, TF, fin_t, Ftfp);
  launch_gemm(st, TF, wtp, w.task_proj.b, nullptr, TP, cB * cU * cT, Ftfp, cE, 0, 0, 0);
  bcast_q_k<<<(cB * cU * cT * cE + 255) / 256, 256, 0, st>>>(U2, QB);
  run_mha(st, w.task_attn, QB, cU * cT, TP, cU * cT, nullptr, CR, ws);
  mean_t_k<<<(cB * cU * cE + 255) / 256, 256, 0, st>>>(CR, CM);

  launch_gemm(st, CM, w.ffn1.w, w.ffn1.b, nullptr, FF1, cB * cU, cE, 2 * cE, 1, 0, 0);
  launch_gemm(st, FF1, w.ffn2.w, w.ffn2.b, CM, PRE, cB * cU, 2 * cE, cE, 0, 0, 0);
  ln_k<<<cB * cU, cE, 0, st>>>(PRE, w.ln_g, w.ln_b, uout, cE);
}

// tA(Kt)/wtp: padded task_proj operands for this layer.
static void task_layer(hipStream_t st, const TaskW& w,
                       const float* tA, int Kt, const float* wtp,
                       const float* u, const float* task_pos_feat, float* ws, float* tout)
{
  float* TW = ws + F_TW; float* T2 = ws + F_T2; float* UL = ws + F_UP;
  float* T3 = ws + F_TW; // reuse after self-attn done
  float* FF1 = ws + F_F1; float* FF2 = ws + F_F2; float* PRE = ws + F_PRE;

  launch_gemm(st, tA, wtp, w.task_proj.b, nullptr, TW, cB * cT, Kt, cE, 0, 0, 0);
  posenc_k<<<cB * cT, cE, 0, st>>>(TW, task_pos_feat, 4);
  run_mha(st, w.self_attn, TW, cT, TW, cT, ws + F_MASK, T2, ws);

  launch_gemm(st, u, w.usv_proj.w, w.usv_proj.b, nullptr, UL, cB * cU, cE, cE, 0, 0, 0);
  run_mha(st, w.usv_attn, T2, cT, UL, cU, nullptr, T3, ws);

  launch_gemm(st, T3, w.ffn1.w, w.ffn1.b, nullptr, FF1, cB * cT, cE, 4 * cE, 1, 0, 0);
  launch_gemm(st, FF1, w.ffn2.w, w.ffn2.b, nullptr, FF2, cB * cT, 4 * cE, 2 * cE, 1, 0, 0);
  launch_gemm(st, FF2, w.ffn3.w, w.ffn3.b, T3, PRE, cB * cT, 2 * cE, cE, 0, 0, 0);
  ln_k<<<cB * cT, cE, 0, st>>>(PRE, w.ln_g, w.ln_b, tout, cE);
}

extern "C" void kernel_launch(void* const* d_in, const int* in_sizes, int n_in,
                              void* d_out, int out_size, void* d_ws, size_t ws_size,
                              hipStream_t stream)
{
  (void)in_sizes; (void)n_in; (void)out_size; (void)ws_size;
  const float* usv_feat  = (const float*)d_in[0];
  const float* task_feat = (const float*)d_in[1];
  const float* edges     = (const float*)d_in[2];

  // ---- parse params (jax pytree flatten: dicts by sorted key, lists in order)
  int cur = 3;
  auto nxt = [&]() { return (const float*)d_in[cur++]; };
  auto lin = [&]() { LinW l; l.b = nxt(); l.w = nxt(); return l; };
  auto mha = [&]() {
    MhaW m; m.ln_b = nxt(); m.ln_g = nxt(); m.wk = nxt();
    m.wo_b = nxt(); m.wo_w = nxt(); m.wq = nxt(); m.wv = nxt(); return m;
  };
  // 'enh'
  LinW enh_l1 = lin(); LinW enh_l2 = lin();
  // 'pool'
  LinW p_a1 = lin(); LinW p_a2 = lin();
  const float* p_ln_b = nxt(); const float* p_ln_g = nxt();
  LinW p_out = lin();
  // 'task' (6 layers)
  TaskW tw[6];
  for (int i = 0; i < 6; ++i) {
    tw[i].ffn1 = lin(); tw[i].ffn2 = lin(); tw[i].ffn3 = lin();
    tw[i].ln_b = nxt(); tw[i].ln_g = nxt();
    tw[i].self_attn = mha(); tw[i].task_proj = lin();
    tw[i].usv_attn = mha(); tw[i].usv_proj = lin();
  }
  // 'usv' (6 layers)
  UsvW uw[6];
  for (int i = 0; i < 6; ++i) {
    uw[i].ffn1 = lin(); uw[i].ffn2 = lin();
    uw[i].ln_b = nxt(); uw[i].ln_g = nxt();
    uw[i].self_attn = mha(); uw[i].task_attn = mha();
    uw[i].task_proj = lin(); uw[i].usv_proj = lin();
  }

  float* ws = (float*)d_ws;

  // ---- pad ragged operands so every hot GEMM takes the b128 fast path
  float* PU0  = ws + F_PU0;   // 64 x 32
  float* PT0  = ws + F_PT0;   // 256 x 32
  float* PWU0 = ws + F_PWU0;  // 128 x 32
  float* PWT0 = ws + F_PWT0;  // 128 x 32
  padcopy_k<<<(64 * 32 + 255) / 256, 256, 0, stream>>>(usv_feat, PU0, 64, 3, 32);
  padcopy_k<<<(256 * 32 + 255) / 256, 256, 0, stream>>>(task_feat, PT0, 256, 4, 32);
  padcopy_k<<<(128 * 32 + 255) / 256, 256, 0, stream>>>(uw[0].usv_proj.w, PWU0, 128, 3, 32);
  padcopy_k<<<(128 * 32 + 255) / 256, 256, 0, stream>>>(tw[0].task_proj.w, PWT0, 128, 4, 32);
  float* PWTU[6];
  for (int i = 0; i < 6; ++i) {
    PWTU[i] = ws + F_PWTU + (size_t)i * PWTU_STRIDE;
    const int sc = (i == 0) ? 7 : 131;
    const int dc = (i == 0) ? 32 : 160;
    padcopy_k<<<(128 * dc + 255) / 256, 256, 0, stream>>>(uw[i].task_proj.w, PWTU[i], 128, sc, dc);
  }

  // kNN mask from (constant) task positions
  hipMemsetAsync(ws + F_MASK, 0, (size_t)cB * cT * cT * sizeof(float), stream);
  knn_mask_k<<<(cB * cT + 63) / 64, 64, 0, stream>>>(task_feat, ws + F_MASK);

  // ---- 6 alternating layers
  for (int i = 0; i < 6; ++i) {
    if (i == 0) {
      usv_layer(stream, uw[0], PU0, 32, PWU0, usv_feat, 3,
                task_feat, 4, PWTU[0], 32, edges, ws, ws + F_U);
      task_layer(stream, tw[0], PT0, 32, PWT0, ws + F_U, task_feat, ws, ws + F_T);
    } else {
      usv_layer(stream, uw[i], ws + F_U, cE, uw[i].usv_proj.w, ws + F_U, cE,
                ws + F_T, cE, PWTU[i], 160, edges, ws, ws + F_U);
      task_layer(stream, tw[i], ws + F_T, cE, tw[i].task_proj.w, ws + F_U, task_feat, ws, ws + F_T);
    }
  }

  // ---- pooling + output head
  float* PA = ws + F_PA; float* PS = ws + F_PS; float* GC = ws + F_GC;
  float* G1 = ws + F_G1; float* E1 = ws + F_E1; float* PRE2 = ws + F_PRE2;
  float* GF = ws + F_G;
  launch_gemm(stream, ws + F_U, p_a1.w, p_a1.b, nullptr, PA, cB * cU, cE, cE / 2, 2, 0, 0);
  launch_gemm(stream, PA, p_a2.w, p_a2.b, nullptr, PS, cB * cU, cE / 2, 1, 0, 0, 0);
  pool_k<<<cB, cE, 0, stream>>>(PS, ws + F_U, GC, cU, 0);
  launch_gemm(stream, ws + F_T, p_a1.w, p_a1.b, nullptr, PA, cB * cT, cE, cE / 2, 2, 0, 0);
  launch_gemm(stream, PA, p_a2.w, p_a2.b, nullptr, PS, cB * cT, cE / 2, 1, 0, 0, 0);
  pool_k<<<cB, cE, 0, stream>>>(PS, ws + F_T, GC, cT, cE);

  launch_gemm(stream, GC, p_out.w, p_out.b, nullptr, PRE2, cB, 2 * cE, 2 * cE, 0, 0, 0);
  ln_k<<<cB, 2 * cE, 0, stream>>>(PRE2, p_ln_g, p_ln_b, G1, 2 * cE);
  launch_gemm(stream, G1, enh_l1.w, enh_l1.b, nullptr, E1, cB, 2 * cE, 2 * cE, 1, 0, 0);
  launch_gemm(stream, E1, enh_l2.w, enh_l2.b, G1, GF, cB, 2 * cE, 2 * cE, 0, 0, 0);

  // ---- gather outputs: (u, t, g) concatenated flat
  float* outp = (float*)d_out;
  hipMemcpyAsync(outp,                 ws + F_U, (size_t)8192  * sizeof(float), hipMemcpyDeviceToDevice, stream);
  hipMemcpyAsync(outp + 8192,          ws + F_T, (size_t)32768 * sizeof(float), hipMemcpyDeviceToDevice, stream);
  hipMemcpyAsync(outp + 8192 + 32768,  GF,       (size_t)1024  * sizeof(float), hipMemcpyDeviceToDevice, stream);
}
